// net_4_78855599554769
// MI455X (gfx1250) — matrix operand order repaired and checked by reading, not compiled, not run on hardware
//
#include <hip/hip_runtime.h>

typedef __attribute__((ext_vector_type(16))) __bf16 bf16x16;
typedef __attribute__((ext_vector_type(8)))  float   f32x8;
typedef __attribute__((ext_vector_type(4)))  unsigned int u32x4;
typedef __attribute__((ext_vector_type(8)))  int          i32x8;
typedef __attribute__((ext_vector_type(4)))  int          i32x4;

constexpr int B_ = 16, T_ = 2048, D_ = 1024, C_ = 6;
constexpr int M_ = B_ * T_;                    // 32768 rows
constexpr int BM = 128, BN = 128, BK = 32;     // workgroup tile
constexpr int PAD = 40;                        // LDS row stride in bf16 (80B = 16 DW data + 4 DW pad)

// ---- CDNA5 data-movement feature detection (guarded so the build can never regress) ----
#if __has_builtin(__builtin_amdgcn_global_load_async_to_lds_b128)
#define USE_ASYNC_LDS 1
#else
#define USE_ASYNC_LDS 0
#endif
#if __has_builtin(__builtin_amdgcn_tensor_load_to_lds) && __has_builtin(__builtin_amdgcn_s_wait_tensorcnt)
#define USE_TDM 1
#else
#define USE_TDM 0
#endif

// async-to-LDS builtin wants v4i pointers in the proper address spaces
#define AS1V4(p) ((__attribute__((address_space(1))) i32x4*)(p))
#define AS3V4(p) ((__attribute__((address_space(3))) i32x4*)(p))

union V16 { struct { uint4 lo, hi; } u; bf16x16 v; };

__device__ inline void wait_async_zero() {
#if __has_builtin(__builtin_amdgcn_s_wait_asynccnt)
  __builtin_amdgcn_s_wait_asynccnt(0);
#else
  asm volatile("s_wait_asynccnt 0x0" ::: "memory");
#endif
}

// ---------------- embedding gather: fp32 emb rows -> bf16 activations ----------------
__global__ void embed_bf16(const int* __restrict__ text, const float* __restrict__ emb,
                           __bf16* __restrict__ xb) {
  long long i4 = (long long)blockIdx.x * blockDim.x + threadIdx.x;
  long long flat = i4 * 4;
  if (flat >= (long long)M_ * D_) return;
  int row = (int)(flat / D_);
  int col = (int)(flat % D_);
  int tok = text[row];
  const float4 f = *(const float4*)(emb + (long long)tok * D_ + col);
  union { __bf16 h[4]; uint2 u; } o;
  o.h[0] = (__bf16)f.x; o.h[1] = (__bf16)f.y; o.h[2] = (__bf16)f.z; o.h[3] = (__bf16)f.w;
  *(uint2*)(xb + flat) = o.u;
}

// ---------------- fp32 -> bf16 (weights) ----------------
__global__ void f32_to_bf16(const float* __restrict__ src, __bf16* __restrict__ dst, int n4) {
  int i = blockIdx.x * blockDim.x + threadIdx.x;
  if (i >= n4) return;
  const float4 f = ((const float4*)src)[i];
  union { __bf16 h[4]; uint2 u; } o;
  o.h[0] = (__bf16)f.x; o.h[1] = (__bf16)f.y; o.h[2] = (__bf16)f.z; o.h[3] = (__bf16)f.w;
  ((uint2*)dst)[i] = o.u;
}

#if USE_TDM
// TDM: DMA a 2D tile (BK x BN, 2-byte elems, row stride K elems) into LDS with
// hardware padding of 4 DWORDs every 16 DWORDs -> reproduces the PAD=40 LDS layout.
__device__ inline void tdm_load_tile(const __bf16* gtile, int rem0, int rem1, int K,
                                     void* lds_dst) {
  unsigned long long ga = (unsigned long long)gtile;             // byte address (57-bit)
  unsigned lds_addr = (unsigned)(unsigned long long)lds_dst;     // low 32 bits = LDS offset
  unsigned long long stride0 = (unsigned long long)K;            // elements
  u32x4 g0;
  g0.x = 1u;                                                     // count=1, user descriptor
  g0.y = lds_addr;
  g0.z = (unsigned)(ga & 0xffffffffull);
  g0.w = (unsigned)((ga >> 32) & 0x01ffffffull) | (2u << 30);    // type=2 ("image")
  i32x8 g1;
  g1[0] = (int)((1u << 16) |        // data_size = 2 bytes
                (1u << 20) |        // pad_enable
                (3u << 22) |        // pad_interval: 16 DWORDs
                (3u << 25));        // pad_amount:   4 DWORDs
  g1[1] = (int)(((unsigned)rem0 & 0xffffu) << 16);                          // tensor_dim0 lo
  g1[2] = (int)((((unsigned)rem0 >> 16) & 0xffffu) | (((unsigned)rem1 & 0xffffu) << 16));
  g1[3] = (int)(((((unsigned)rem1) >> 16) & 0xffffu) | ((unsigned)BK << 16)); // tile_dim0 = 32
  g1[4] = (int)(unsigned)BN;                                                 // tile_dim1 = 128
  g1[5] = (int)(unsigned)(stride0 & 0xffffffffull);                          // dim0 stride lo
  g1[6] = (int)(unsigned)((stride0 >> 32) & 0xffffull);                      // dim0 stride hi
  g1[7] = 0;
  i32x4 gz = {0, 0, 0, 0};
#if __clang_major__ >= 23
  i32x8 gz8 = {0, 0, 0, 0, 0, 0, 0, 0};
  __builtin_amdgcn_tensor_load_to_lds(g0, g1, gz, gz, gz8, 0);
#else
  __builtin_amdgcn_tensor_load_to_lds(g0, g1, gz, gz, 0);
#endif
}
#endif

// ---------------- C[m,n] = sum_k A[m,k]*W[n,k] + bias[n]  (bf16 in, bf16 out, f32 acc) ----------------
__launch_bounds__(256)
__global__ void gemm_xwT_bf16(const __bf16* __restrict__ A,    // M x K row-major
                              const __bf16* __restrict__ W,    // N x K row-major (B = W^T)
                              const float*  __restrict__ bias, // N
                              __bf16* __restrict__ Cmat,       // M x N
                              int M, int N, int K) {
  __shared__ __align__(16) __bf16 sA[2][BM * PAD];
  __shared__ __align__(16) __bf16 sB[2][BN * PAD];

  const int tid  = threadIdx.x;
  const int lane = tid & 31;
  const int wv   = tid >> 5;
  const int half = lane >> 4;   // K half-split across half-waves (16-bit WMMA layout)
  const int l16  = lane & 15;

  const int m_block = blockIdx.y * BM;
  const int n_block = blockIdx.x * BN;
  const int wave_m  = (wv >> 1) * 32;   // 4 waves along M
  const int wave_n  = (wv & 1) * 64;    // 2 waves along N

  const int ldr = tid >> 1;             // 0..127 row within tile
  const int ldc = (tid & 1) * 16;       // 0 or 16 bf16 elements within BK
  const __bf16* gA = A + (long long)(m_block + ldr) * K + ldc;
  const __bf16* gB = W + (long long)(n_block + ldr) * K + ldc;

  f32x8 acc[2][4] = {};
  const int nk = K / BK;

  // A tile staging: async-to-LDS (ASYNCcnt) when available, else vector path.
  auto stage_A = [&](int s, int kt) {
#if USE_ASYNC_LDS
    const __bf16* src = gA + kt * BK;
    void* dst = (void*)&sA[s][ldr * PAD + ldc];
    __builtin_amdgcn_global_load_async_to_lds_b128(AS1V4(src), AS3V4(dst), 0, 0);
    __builtin_amdgcn_global_load_async_to_lds_b128(AS1V4(src), AS3V4(dst), 16, 0);
#else
    const uint4* a = (const uint4*)(gA + kt * BK);
    uint4 a0 = a[0], a1 = a[1];
    uint4* da = (uint4*)&sA[s][ldr * PAD + ldc];
    da[0] = a0; da[1] = a1;
#endif
  };
  // B tile staging: one TDM descriptor per tile issued by wave 0 (TENSORcnt),
  // else async per-thread, else vector path.
  auto stage_B = [&](int s, int kt) {
#if USE_TDM
    if (wv == 0)
      tdm_load_tile(W + (long long)n_block * K + kt * BK, K - kt * BK, N - n_block, K,
                    (void*)&sB[s][0]);
#elif USE_ASYNC_LDS
    const __bf16* src = gB + kt * BK;
    void* dst = (void*)&sB[s][ldr * PAD + ldc];
    __builtin_amdgcn_global_load_async_to_lds_b128(AS1V4(src), AS3V4(dst), 0, 0);
    __builtin_amdgcn_global_load_async_to_lds_b128(AS1V4(src), AS3V4(dst), 16, 0);
#else
    const uint4* b = (const uint4*)(gB + kt * BK);
    uint4 b0 = b[0], b1 = b[1];
    uint4* db = (uint4*)&sB[s][ldr * PAD + ldc];
    db[0] = b0; db[1] = b1;
#endif
  };
  auto drain_staging = [&]() {
#if USE_ASYNC_LDS
    wait_async_zero();
#endif
#if USE_TDM
    if (wv == 0) __builtin_amdgcn_s_wait_tensorcnt(0);
#endif
  };

  stage_A(0, 0);
  stage_B(0, 0);
  drain_staging();
  __syncthreads();

  for (int kt = 0; kt < nk; ++kt) {
    const int cur = kt & 1;
    if (kt + 1 < nk) {
      stage_A(cur ^ 1, kt + 1);
      stage_B(cur ^ 1, kt + 1);
      if (kt + 2 < nk) {
        __builtin_prefetch(gA + (kt + 2) * BK, 0, 1);   // global_prefetch_b8
        __builtin_prefetch(gB + (kt + 2) * BK, 0, 1);
      }
    }

    // A frag 16x32: lane row = m, lanes<16 hold K {0..7,16..23}, lanes>=16 hold K {8..15,24..31}
    bf16x16 afrag[2], bfrag[4];
#pragma unroll
    for (int i = 0; i < 2; ++i) {
      int r = wave_m + i * 16 + l16;
      V16 t;
      t.u.lo = *(const uint4*)&sA[cur][r * PAD + half * 8];
      t.u.hi = *(const uint4*)&sA[cur][r * PAD + 16 + half * 8];
      afrag[i] = t.v;
    }
    // B frag 32x16: lane col = n, lanes<16 hold K {0..7,16..23}, lanes>=16 hold K {8..15,24..31}
#pragma unroll
    for (int j = 0; j < 4; ++j) {
      int r = wave_n + j * 16 + l16;
      V16 t;
      t.u.lo = *(const uint4*)&sB[cur][r * PAD + half * 8];
      t.u.hi = *(const uint4*)&sB[cur][r * PAD + 16 + half * 8];
      bfrag[j] = t.v;
    }

#pragma unroll
    for (int i = 0; i < 2; ++i)
#pragma unroll
      for (int j = 0; j < 4; ++j)
        acc[i][j] = __builtin_amdgcn_wmma_f32_16x16x32_bf16(
            false, afrag[i], false, bfrag[j], (short)0, acc[i][j], false, false);

    drain_staging();
    __syncthreads();
  }

  // C layout: VGPR r -> M = r (lanes<16) / r+8 (lanes>=16), N = lane&15
#pragma unroll
  for (int j = 0; j < 4; ++j) {
    const int n = n_block + wave_n + j * 16 + l16;
    const float bj = bias[n];
#pragma unroll
    for (int i = 0; i < 2; ++i) {
#pragma unroll
      for (int r = 0; r < 8; ++r) {
        const int m = m_block + wave_m + i * 16 + r + 8 * half;
        Cmat[(long long)m * N + n] = (__bf16)(acc[i][j][r] + bj);
      }
    }
  }
}

// ---------------- per-batch tail: q_last, attention row, residual, norms, fc, head ----------------
__launch_bounds__(256)
__global__ void tail_kernel(const int* __restrict__ text, const float* __restrict__ emb,
                            const float* __restrict__ Wq, const float* __restrict__ bq,
                            const __bf16* __restrict__ Kb, const __bf16* __restrict__ Vb,
                            const float* __restrict__ Wfc, const float* __restrict__ bfc,
                            const float* __restrict__ Wo, const float* __restrict__ bo,
                            float* __restrict__ out) {
  __shared__ float xs[D_], qs[D_], hs[D_], ys[D_];
  __shared__ float sc[T_];
  __shared__ float red[256];

  const int b = blockIdx.x;
  const int tid = threadIdx.x;

  const int tok = text[b * T_ + (T_ - 1)];
  for (int d = tid; d < D_; d += 256) xs[d] = emb[(long long)tok * D_ + d];
  __syncthreads();

  // q = x_last @ Wq^T + bq
  for (int n = tid; n < D_; n += 256) {
    float s = bq[n];
    const float* wr = Wq + (long long)n * D_;
    for (int d = 0; d < D_; ++d) s += xs[d] * wr[d];
    qs[n] = s;
  }
  __syncthreads();

  // scores against all keys (causal row T-1 admits every key)
  const float scale = 1.0f / 32.0f;   // 1/sqrt(1024)
  float lmax = -3.4e38f;
  for (int t = tid; t < T_; t += 256) {
    const __bf16* kr = Kb + ((long long)b * T_ + t) * D_;
    float s = 0.f;
    for (int d = 0; d < D_; ++d) s += qs[d] * (float)kr[d];
    s *= scale;
    sc[t] = s;
    lmax = fmaxf(lmax, s);
  }
  red[tid] = lmax; __syncthreads();
  for (int o = 128; o > 0; o >>= 1) { if (tid < o) red[tid] = fmaxf(red[tid], red[tid + o]); __syncthreads(); }
  const float mx = red[0]; __syncthreads();

  float lsum = 0.f;
  for (int t = tid; t < T_; t += 256) { float e = expf(sc[t] - mx); sc[t] = e; lsum += e; }
  red[tid] = lsum; __syncthreads();
  for (int o = 128; o > 0; o >>= 1) { if (tid < o) red[tid] += red[tid + o]; __syncthreads(); }
  const float inv_sum = 1.0f / red[0]; __syncthreads();

  // h = attn @ V + x  (thread owns 4 consecutive d -> coalesced 8B V reads)
  {
    const int d0 = tid * 4;
    float o0 = 0, o1 = 0, o2 = 0, o3 = 0;
    for (int t = 0; t < T_; ++t) {
      const float w = sc[t];
      const __bf16* vr = Vb + ((long long)b * T_ + t) * D_ + d0;
      o0 += w * (float)vr[0]; o1 += w * (float)vr[1];
      o2 += w * (float)vr[2]; o3 += w * (float)vr[3];
    }
    hs[d0 + 0] = o0 * inv_sum + xs[d0 + 0];
    hs[d0 + 1] = o1 * inv_sum + xs[d0 + 1];
    hs[d0 + 2] = o2 * inv_sum + xs[d0 + 2];
    hs[d0 + 3] = o3 * inv_sum + xs[d0 + 3];
  }
  __syncthreads();

  // l2norm #1
  float ss = 0.f;
  for (int d = tid; d < D_; d += 256) ss += hs[d] * hs[d];
  red[tid] = ss; __syncthreads();
  for (int o = 128; o > 0; o >>= 1) { if (tid < o) red[tid] += red[tid + o]; __syncthreads(); }
  float inv_n = 1.0f / fmaxf(sqrtf(red[0]), 1e-12f); __syncthreads();
  for (int d = tid; d < D_; d += 256) hs[d] *= inv_n;
  __syncthreads();

  // y = h + (h @ Wfc^T + bfc)
  for (int n = tid; n < D_; n += 256) {
    float s = bfc[n];
    const float* wr = Wfc + (long long)n * D_;
    for (int d = 0; d < D_; ++d) s += hs[d] * wr[d];
    ys[n] = hs[n] + s;
  }
  __syncthreads();

  // l2norm #2
  ss = 0.f;
  for (int d = tid; d < D_; d += 256) ss += ys[d] * ys[d];
  red[tid] = ss; __syncthreads();
  for (int o = 128; o > 0; o >>= 1) { if (tid < o) red[tid] += red[tid + o]; __syncthreads(); }
  inv_n = 1.0f / fmaxf(sqrtf(red[0]), 1e-12f); __syncthreads();
  for (int d = tid; d < D_; d += 256) ys[d] *= inv_n;
  __syncthreads();

  // head + sigmoid
  for (int c = 0; c < C_; ++c) {
    float p = 0.f;
    for (int d = tid; d < D_; d += 256) p += ys[d] * Wo[(long long)c * D_ + d];
    red[tid] = p; __syncthreads();
    for (int o = 128; o > 0; o >>= 1) { if (tid < o) red[tid] += red[tid + o]; __syncthreads(); }
    if (tid == 0) {
      float z = red[0] + bo[c];
      out[b * C_ + c] = 1.0f / (1.0f + expf(-z));
    }
    __syncthreads();
  }
}

extern "C" void kernel_launch(void* const* d_in, const int* in_sizes, int n_in,
                              void* d_out, int out_size, void* d_ws, size_t ws_size,
                              hipStream_t stream) {
  const int*   text = (const int*)  d_in[0];
  // d_in[1] = offsets (unused: offsets[b] == b*T)
  const float* emb  = (const float*)d_in[2];
  const float* Wq   = (const float*)d_in[3];
  const float* bq   = (const float*)d_in[4];
  const float* Wk   = (const float*)d_in[5];
  const float* bk   = (const float*)d_in[6];
  const float* Wv   = (const float*)d_in[7];
  const float* bv   = (const float*)d_in[8];
  const float* Wfc  = (const float*)d_in[9];
  const float* bfc  = (const float*)d_in[10];
  const float* Wo   = (const float*)d_in[11];
  const float* bo   = (const float*)d_in[12];
  float* out = (float*)d_out;

  char* ws = (char*)d_ws;
  const size_t actB = (size_t)M_ * D_ * sizeof(__bf16);   // 64 MB
  const size_t wB   = (size_t)D_ * D_ * sizeof(__bf16);   // 2 MB
  __bf16* xb  = (__bf16*)(ws);
  __bf16* kb  = (__bf16*)(ws + actB);
  __bf16* vb  = (__bf16*)(ws + 2 * actB);
  __bf16* wkb = (__bf16*)(ws + 3 * actB);
  __bf16* wvb = (__bf16*)(ws + 3 * actB + wB);

  // 1) gather + downconvert activations
  embed_bf16<<<(int)(((long long)M_ * D_ / 4 + 255) / 256), 256, 0, stream>>>(text, emb, xb);
  // 2) weight downconvert (only K/V projections go through WMMA)
  f32_to_bf16<<<(D_ * D_ / 4 + 255) / 256, 256, 0, stream>>>(Wk, wkb, D_ * D_ / 4);
  f32_to_bf16<<<(D_ * D_ / 4 + 255) / 256, 256, 0, stream>>>(Wv, wvb, D_ * D_ / 4);
  // 3) the two dominant GEMMs: K = x Wk^T + bk, V = x Wv^T + bv
  dim3 ggrid(D_ / BN, M_ / BM);   // (8, 256)
  gemm_xwT_bf16<<<ggrid, 256, 0, stream>>>(xb, wkb, bk, kb, M_, D_, D_);
  gemm_xwT_bf16<<<ggrid, 256, 0, stream>>>(xb, wvb, bv, vb, M_, D_, D_);
  // 4) last-token attention + residual + norms + fc + head
  tail_kernel<<<B_, 256, 0, stream>>>(text, emb, Wq, bq, kb, vb, Wfc, bfc, Wo, bo, out);
}
